// EvolvingGNN_83614423318998
// MI455X (gfx1250) — matrix operand order repaired and checked by reading, not compiled, not run on hardware
//
#include <hip/hip_runtime.h>
#include <hip/hip_bf16.h>

typedef __attribute__((ext_vector_type(16))) _Float16 v16h;
typedef __attribute__((ext_vector_type(8)))  float    v8f;

#define TSTEPS 5
#define DIM 64

// ---------------------------------------------------------------- utilities

__device__ __forceinline__ void atomAddF(float* p, float v) {
    __hip_atomic_fetch_add(p, v, __ATOMIC_RELAXED, __HIP_MEMORY_SCOPE_AGENT);
}

// Load the two contiguous 8-float runs [p..p+7] and [p+16..p+23] that one lane
// owns in a 16x32 f16 A-fragment, applying ReLU, convert to f16.
__device__ __forceinline__ v16h pack_a_relu(const float* __restrict__ p) {
    float4 f0 = *(const float4*)(p);
    float4 f1 = *(const float4*)(p + 4);
    float4 f2 = *(const float4*)(p + 16);
    float4 f3 = *(const float4*)(p + 20);
    v16h a;
    a[0]  = (_Float16)fmaxf(f0.x, 0.f); a[1]  = (_Float16)fmaxf(f0.y, 0.f);
    a[2]  = (_Float16)fmaxf(f0.z, 0.f); a[3]  = (_Float16)fmaxf(f0.w, 0.f);
    a[4]  = (_Float16)fmaxf(f1.x, 0.f); a[5]  = (_Float16)fmaxf(f1.y, 0.f);
    a[6]  = (_Float16)fmaxf(f1.z, 0.f); a[7]  = (_Float16)fmaxf(f1.w, 0.f);
    a[8]  = (_Float16)fmaxf(f2.x, 0.f); a[9]  = (_Float16)fmaxf(f2.y, 0.f);
    a[10] = (_Float16)fmaxf(f2.z, 0.f); a[11] = (_Float16)fmaxf(f2.w, 0.f);
    a[12] = (_Float16)fmaxf(f3.x, 0.f); a[13] = (_Float16)fmaxf(f3.y, 0.f);
    a[14] = (_Float16)fmaxf(f3.z, 0.f); a[15] = (_Float16)fmaxf(f3.w, 0.f);
    return a;
}

__device__ __forceinline__ v16h pack_a(const float* __restrict__ p) {
    float4 f0 = *(const float4*)(p);
    float4 f1 = *(const float4*)(p + 4);
    float4 f2 = *(const float4*)(p + 16);
    float4 f3 = *(const float4*)(p + 20);
    v16h a;
    a[0]  = (_Float16)f0.x; a[1]  = (_Float16)f0.y;
    a[2]  = (_Float16)f0.z; a[3]  = (_Float16)f0.w;
    a[4]  = (_Float16)f1.x; a[5]  = (_Float16)f1.y;
    a[6]  = (_Float16)f1.z; a[7]  = (_Float16)f1.w;
    a[8]  = (_Float16)f2.x; a[9]  = (_Float16)f2.y;
    a[10] = (_Float16)f2.z; a[11] = (_Float16)f2.w;
    a[12] = (_Float16)f3.x; a[13] = (_Float16)f3.y;
    a[14] = (_Float16)f3.z; a[15] = (_Float16)f3.w;
    return a;
}

// ------------------------------------------------ 1) workspace init per call

__global__ void egnn_init(float* __restrict__ agg, float* __restrict__ deg,
                          long long n64, int N) {
    long long i = (long long)blockIdx.x * blockDim.x + threadIdx.x;
    if (i < n64) agg[i] = 0.0f;
    if (i < N)   deg[i] = 1.0f;   // self-loop contributes exactly 1 per node
}

// ------------------------------------------------ 2) degree count + rsqrt

__global__ void egnn_deg(const int* __restrict__ ei, float* __restrict__ deg,
                         long long E) {
    long long e = (long long)blockIdx.x * blockDim.x + threadIdx.x;
    if (e < E) atomAddF(&deg[ei[E + e]], 1.0f);   // col = edge_index[1]
}

__global__ void egnn_dinv(float* __restrict__ deg, int N) {
    int i = blockIdx.x * blockDim.x + threadIdx.x;
    if (i < N) deg[i] = rsqrtf(deg[i]);           // deg >= 1 always
}

// ------------------------------------------------ 3) LSTM weight evolution
// One workgroup, 5 sequential steps on 64x64 matrices. gates = w^T @ w_ih^T
// + h @ w_hh^T ; PyTorch gate order i,f,g,o.  Final w written to wfin.

__global__ void __launch_bounds__(256)
egnn_lstm(const float* __restrict__ init_w,
          const float* __restrict__ w_ih, const float* __restrict__ w_hh,
          const float* __restrict__ b_ih, const float* __restrict__ b_hh,
          float* __restrict__ wfin) {
    __shared__ float ws_[64 * 64];   // ws_[k*64+r] = w[k][r]
    __shared__ float hs_[64 * 64];   // hs_[r*64+k] = h[r][k]
    __shared__ float cs_[64 * 64];
    int tid = threadIdx.x;
    for (int i = tid; i < 4096; i += 256) {
        ws_[i] = init_w[i];
        hs_[i] = 0.0f;
        cs_[i] = 0.0f;
    }
    __syncthreads();
    for (int step = 0; step < TSTEPS; ++step) {
        float hv[16], cv[16];
        #pragma unroll 1
        for (int ii = 0; ii < 16; ++ii) {
            int idx = tid + ii * 256;
            int r = idx >> 6, k = idx & 63;
            float gi = b_ih[k]       + b_hh[k];
            float gf = b_ih[64 + k]  + b_hh[64 + k];
            float gg = b_ih[128 + k] + b_hh[128 + k];
            float go = b_ih[192 + k] + b_hh[192 + k];
            for (int kk = 0; kk < 64; ++kk) {
                float xb = ws_[kk * 64 + r];     // xb[r][kk] = w[kk][r]
                float hh = hs_[r * 64 + kk];
                gi += xb * w_ih[k * 64 + kk]         + hh * w_hh[k * 64 + kk];
                gf += xb * w_ih[(64 + k) * 64 + kk]  + hh * w_hh[(64 + k) * 64 + kk];
                gg += xb * w_ih[(128 + k) * 64 + kk] + hh * w_hh[(128 + k) * 64 + kk];
                go += xb * w_ih[(192 + k) * 64 + kk] + hh * w_hh[(192 + k) * 64 + kk];
            }
            float iv = 1.0f / (1.0f + __expf(-gi));
            float fv = 1.0f / (1.0f + __expf(-gf));
            float gv = tanhf(gg);
            float ov = 1.0f / (1.0f + __expf(-go));
            float cc = fv * cs_[r * 64 + k] + iv * gv;
            cv[ii] = cc;
            hv[ii] = ov * tanhf(cc);
        }
        __syncthreads();
        #pragma unroll 1
        for (int ii = 0; ii < 16; ++ii) {
            int idx = tid + ii * 256;
            int r = idx >> 6, k = idx & 63;
            cs_[r * 64 + k] = cv[ii];
            hs_[r * 64 + k] = hv[ii];
            ws_[k * 64 + r] = hv[ii];            // new w = h^T
        }
        __syncthreads();
    }
    for (int i = tid; i < 4096; i += 256) wfin[i] = ws_[i];
}

// ------------------------------------------------ 4) xw = xs[T-1] @ w_final
// WMMA f16: per wave 16 rows, K=64 (2 chunks of 32), 4 column tiles of 16.

__global__ void __launch_bounds__(256)
egnn_gemm_xw(const float* __restrict__ x, const float* __restrict__ w,
             float* __restrict__ xw, int N) {
    __shared__ _Float16 wt[64 * 64];             // wt[n*64+k] = w[k with bits 3,4 swapped][n] (f16)
    int tid = threadIdx.x;
    for (int i = tid; i < 64 * 64; i += 256) {
        int n = i >> 6, k = i & 63;
        wt[n * 64 + k] = (_Float16)w[((k & ~24) | ((k & 8) << 1) | ((k & 16) >> 1)) * 64 + n];
    }
    __syncthreads();
    int wave = tid >> 5, lane = tid & 31;
    int tile = blockIdx.x * 8 + wave;
    int ntiles = (N + 15) >> 4;
    if (tile >= ntiles) return;                  // wave-uniform: EXEC stays full
    int row_base = tile << 4;
    bool full = (row_base + 16 <= N);            // wave-uniform fast path
    int m = lane & 15, half = lane >> 4;
    int row = row_base + m;
    if (row >= N) row = N - 1;                   // clamp; stores are masked
    v16h a0 = pack_a(x + (size_t)row * 64 + half * 8);
    v16h a1 = pack_a(x + (size_t)row * 64 + 32 + half * 8);
    #pragma unroll
    for (int t = 0; t < 4; ++t) {
        v8f c = {};
        const v16h* b0 = (const v16h*)&wt[(16 * t + m) * 64 + half * 16];
        const v16h* b1 = (const v16h*)&wt[(16 * t + m) * 64 + 32 + half * 16];
        c = __builtin_amdgcn_wmma_f32_16x16x32_f16(false, a0, false, *b0,
                                                   (short)0, c, false, false);
        c = __builtin_amdgcn_wmma_f32_16x16x32_f16(false, a1, false, *b1,
                                                   (short)0, c, false, false);
        float* outp = xw + ((size_t)(row_base + half * 8)) * 64 + 16 * t + m;
        if (full) {                              // uniform branch: no exec juggling
            #pragma unroll
            for (int r = 0; r < 8; ++r) outp[(size_t)r * 64] = c[r];
        } else {
            #pragma unroll
            for (int r = 0; r < 8; ++r) {
                if (row_base + half * 8 + r < N) outp[(size_t)r * 64] = c[r];
            }
        }
    }
}

// ------------------------------------------------ 5) scatter: agg[col] += norm*xw[row]
// One thread per (edge-or-selfloop, 4-float group); fp32 global atomics.

__global__ void egnn_scatter(const int* __restrict__ ei, const float* __restrict__ dinv,
                             const float* __restrict__ xw, float* __restrict__ agg,
                             long long E, int N) {
    long long t = (long long)blockIdx.x * blockDim.x + threadIdx.x;
    long long total = (E + (long long)N) * 16;
    if (t >= total) return;
    long long item = t >> 4;
    int g = (int)(t & 15);
    int r, c;
    float nrm;
    if (item < E) {
        r = ei[item];
        c = ei[E + item];
        nrm = dinv[r] * dinv[c];
    } else {
        r = c = (int)(item - E);                 // self loop
        float dv = dinv[r];
        nrm = dv * dv;
    }
    float4 v = *(const float4*)(xw + (size_t)r * 64 + g * 4);
    float* dst = agg + (size_t)c * 64 + g * 4;
    atomAddF(dst + 0, nrm * v.x);
    atomAddF(dst + 1, nrm * v.y);
    atomAddF(dst + 2, nrm * v.z);
    atomAddF(dst + 3, nrm * v.w);
}

// ------------------------------------------------ 6) edge MLP with WMMA
// [E,144] @ [144,64] (K padded to 160 = 5 chunks); ReLU(x_last) fused into the
// A gather; epilogue fuses +b1, ReLU, dot with w2, +b2, shfl-xor reduction.

__global__ void __launch_bounds__(256)
egnn_edge_mlp(const float* __restrict__ agg, const int* __restrict__ ei,
              const float* __restrict__ eattr,
              const float* __restrict__ w1, const float* __restrict__ b1,
              const float* __restrict__ w2, const float* __restrict__ b2,
              float* __restrict__ out, long long E) {
    __shared__ _Float16 wt[64 * 160];            // wt[n*160+k] = w1[k with bits 3,4 swapped][n], zero pad
    __shared__ float w2s[64];
    __shared__ float b1s[64];
    int tid = threadIdx.x;
    for (int i = tid; i < 64 * 160; i += 256) {
        int n = i / 160, k = i - n * 160;
        int kk = (k & ~24) | ((k & 8) << 1) | ((k & 16) >> 1);
        wt[i] = (kk < 144) ? (_Float16)w1[kk * 64 + n] : (_Float16)0.0f;
    }
    if (tid < 64) { w2s[tid] = w2[tid]; b1s[tid] = b1[tid]; }
    __syncthreads();
    int wave = tid >> 5, lane = tid & 31;
    long long e_base = (((long long)blockIdx.x * 8 + wave) << 4);
    if (e_base >= E) return;                     // wave-uniform
    bool full = (e_base + 16 <= E);              // wave-uniform fast path
    int m = lane & 15, half = lane >> 4;
    long long e = e_base + m;
    if (e >= E) e = E - 1;                       // clamp; stores masked
    int src = ei[e];
    int dst = ei[E + e];
    int bo = half * 8;
    const float* ps = agg + (size_t)src * 64;
    const float* pd = agg + (size_t)dst * 64;
    v16h a[5];
    a[0] = pack_a_relu(ps + bo);                 // K   0..31  : relu(h_src lo)
    a[1] = pack_a_relu(ps + 32 + bo);            // K  32..63  : relu(h_src hi)
    a[2] = pack_a_relu(pd + bo);                 // K  64..95  : relu(h_dst lo)
    a[3] = pack_a_relu(pd + 32 + bo);            // K  96..127 : relu(h_dst hi)
    {                                            // K 128..159 : edge_attr | zeros
        const float* pa = eattr + (size_t)e * 16 + bo;
        float4 f0 = *(const float4*)(pa);
        float4 f1 = *(const float4*)(pa + 4);
        v16h av;
        av[0] = (_Float16)f0.x; av[1] = (_Float16)f0.y;
        av[2] = (_Float16)f0.z; av[3] = (_Float16)f0.w;
        av[4] = (_Float16)f1.x; av[5] = (_Float16)f1.y;
        av[6] = (_Float16)f1.z; av[7] = (_Float16)f1.w;
        #pragma unroll
        for (int j = 8; j < 16; ++j) av[j] = (_Float16)0.0f;
        a[4] = av;
    }
    float b2v = b2[0];
    float s[8];
    #pragma unroll
    for (int r = 0; r < 8; ++r) s[r] = 0.0f;
    #pragma unroll
    for (int t = 0; t < 4; ++t) {
        v8f c = {};
        #pragma unroll
        for (int kc = 0; kc < 5; ++kc) {
            const v16h* bp = (const v16h*)&wt[(16 * t + m) * 160 + kc * 32 + half * 16];
            c = __builtin_amdgcn_wmma_f32_16x16x32_f16(false, a[kc], false, *bp,
                                                       (short)0, c, false, false);
        }
        float w2v = w2s[16 * t + m];             // column n = 16t + (lane&15)
        float b1v = b1s[16 * t + m];
        #pragma unroll
        for (int r = 0; r < 8; ++r)
            s[r] += fmaxf(c[r] + b1v, 0.0f) * w2v;
    }
    // reduce the 16 columns held across each 16-lane half (xor stays in-group)
    #pragma unroll
    for (int r = 0; r < 8; ++r) {
        #pragma unroll
        for (int off = 8; off; off >>= 1) s[r] += __shfl_xor(s[r], off, 32);
    }
    if (m == 0) {
        long long eb = e_base + half * 8;        // edge M = r + 8*half
        if (full) {
            #pragma unroll
            for (int r = 0; r < 8; ++r) out[eb + r] = s[r] + b2v;
        } else {
            #pragma unroll
            for (int r = 0; r < 8; ++r) {
                if (eb + r < E) out[eb + r] = s[r] + b2v;
            }
        }
    }
}

// ---------------------------------------------------------------- launcher

extern "C" void kernel_launch(void* const* d_in, const int* in_sizes, int n_in,
                              void* d_out, int out_size, void* d_ws, size_t ws_size,
                              hipStream_t stream) {
    const float* xs     = (const float*)d_in[0];
    const int*   ei     = (const int*)d_in[1];
    const float* eattr  = (const float*)d_in[2];
    const float* init_w = (const float*)d_in[3];
    const float* w_ih   = (const float*)d_in[4];
    const float* w_hh   = (const float*)d_in[5];
    const float* b_ih   = (const float*)d_in[6];
    const float* b_hh   = (const float*)d_in[7];
    const float* mw1    = (const float*)d_in[8];
    const float* mb1    = (const float*)d_in[9];
    const float* mw2    = (const float*)d_in[10];
    const float* mb2    = (const float*)d_in[11];
    float* out = (float*)d_out;

    const int       N = in_sizes[0] / (TSTEPS * DIM);
    const long long E = (long long)in_sizes[1] / 2;

    auto align256 = [](size_t x) { return (x + 255) & ~(size_t)255; };
    char* ws = (char*)d_ws;
    float* wfin = (float*)ws;                               // 64*64 f32
    size_t off  = align256(64 * 64 * sizeof(float));
    float* dinv = (float*)(ws + off);                       // N f32 (deg, then deg^-1/2)
    off += align256((size_t)N * sizeof(float));
    float* xw   = (float*)(ws + off);                       // N*64 f32
    off += align256((size_t)N * 64 * sizeof(float));
    float* agg  = (float*)(ws + off);                       // N*64 f32
    (void)ws_size; (void)n_in; (void)out_size;

    const long long n64 = (long long)N * 64;

    // 1) zero agg, deg := 1 (self loops)
    egnn_init<<<(unsigned)((n64 + 255) / 256), 256, 0, stream>>>(agg, dinv, n64, N);
    // 2) in-degree over col, then dinv = deg^-0.5
    egnn_deg<<<(unsigned)((E + 255) / 256), 256, 0, stream>>>(ei, dinv, E);
    egnn_dinv<<<(unsigned)((N + 255) / 256), 256, 0, stream>>>(dinv, N);
    // 3) evolve LSTM weights (independent of data path above)
    egnn_lstm<<<1, 256, 0, stream>>>(init_w, w_ih, w_hh, b_ih, b_hh, wfin);
    // 4) xw = xs[T-1] @ w_final (WMMA)
    const float* xs_last = xs + (size_t)(TSTEPS - 1) * N * DIM;
    int ntiles = (N + 15) / 16;
    egnn_gemm_xw<<<(unsigned)((ntiles + 7) / 8), 256, 0, stream>>>(xs_last, wfin, xw, N);
    // 5) normalized scatter-add (edges + self loops)
    long long sc_threads = (E + (long long)N) * 16;
    egnn_scatter<<<(unsigned)((sc_threads + 255) / 256), 256, 0, stream>>>(
        ei, dinv, xw, agg, E, N);
    // 6) edge MLP (WMMA) -> logits
    long long etiles = (E + 15) / 16;
    egnn_edge_mlp<<<(unsigned)((etiles + 7) / 8), 256, 0, stream>>>(
        agg, ei, eattr, mw1, mb1, mw2, mb2, out, E);
}
